// HGTConv_14989435863176
// MI455X (gfx1250) — matrix operand order repaired and checked by reading, not compiled, not run on hardware
//
#include <hip/hip_runtime.h>
#include <math.h>

#define NP_   100000
#define NA_   50000
#define EE_   500000
#define DM_   128
#define HH_   8
#define DD_   16
#define NQKV_ 384

typedef __attribute__((ext_vector_type(16))) __bf16 v16bf;
typedef __attribute__((ext_vector_type(8)))  float  v8f;

// ---------------------------------------------------------------- utilities
__global__ void k_fill(float* __restrict__ p, float v, int n) {
    int i = blockIdx.x * blockDim.x + threadIdx.x;
    if (i < n) p[i] = v;
}

__device__ __forceinline__ void atomicMaxF(float* addr, float val) {
    // monotonic int mapping; buffer initialized to -inf (0xFF800000)
    if (val >= 0.0f) atomicMax((int*)addr, __float_as_int(val));
    else             atomicMin((unsigned int*)addr, (unsigned int)__float_as_int(val));
}

// ------------------------------------------------- fused weight construction
// W3[k][n], n in [0,128): Wq; [128,256): Wk @ blockdiag(a_rel); [256,384): Wv @ blockdiag(m_rel)
__global__ void k_build_w3(const float* __restrict__ Wq, const float* __restrict__ bq,
                           const float* __restrict__ Wk, const float* __restrict__ bk,
                           const float* __restrict__ Wv, const float* __restrict__ bv,
                           const float* __restrict__ aR, const float* __restrict__ mR,
                           float* __restrict__ W3, float* __restrict__ b3) {
    int n = blockIdx.x;          // 0..383
    int k = threadIdx.x;         // 0..127
    float val, bval;
    if (n < DM_) {
        val  = Wq[k * DM_ + n];
        bval = bq[n];
    } else {
        int nn = n & (DM_ - 1);
        int h = nn >> 4, e = nn & 15;
        const float* W = (n < 2 * DM_) ? Wk : Wv;
        const float* b = (n < 2 * DM_) ? bk : bv;
        const float* R = (n < 2 * DM_) ? aR : mR;
        float s = 0.f, sb = 0.f;
#pragma unroll
        for (int d = 0; d < DD_; ++d) {
            float r = R[(h * DD_ + d) * DD_ + e];
            s  += W[k * DM_ + h * DD_ + d] * r;
            sb += b[h * DD_ + d] * r;
        }
        val = s; bval = sb;
    }
    W3[k * NQKV_ + n] = val;
    if (k == 0) b3[n] = bval;
}

// --------------------------- pack fp32 weight into wave32 WMMA B fragments
// B 32x16 bf16 layout: kbase=(lane>>4)*8; slots 0..7 -> K=kbase+j, slots 8..15 -> K=16+kbase+j
__global__ void k_pack_w(const float* __restrict__ W, int NC,
                         __bf16* __restrict__ hiout, __bf16* __restrict__ loout) {
    int t = blockIdx.x * blockDim.x + threadIdx.x;
    if (t >= DM_ * NC) return;
    int k = t / NC, n = t % NC;
    float f = W[(size_t)k * NC + n];
    __bf16 h = (__bf16)f;
    __bf16 l = (__bf16)(f - (float)h);
    int chunk = k >> 5, kin = k & 31;
    int ntile = n >> 4, ncol = n & 15;
    int lane = ncol + 16 * ((kin >> 3) & 1);
    int slot = (kin & 7) + 8 * (kin >> 4);
    int NT = NC >> 4;
    size_t idx = (((size_t)(chunk * NT + ntile) * 32) + lane) * 16 + slot;
    hiout[idx] = h;
    loout[idx] = l;
}

// ------------------------------------------- bf16x3 WMMA GEMM (K=128 fixed)
// Out[16 rows x NC] per block; 4 waves split the NC/16 column tiles.
// If skip != nullptr: Out = sigmoid(skip)*(acc+bias) + (1-sigmoid(skip))*Xres
__global__ __launch_bounds__(128) void k_gemm_bf16x3(
        const float* __restrict__ X, const __bf16* __restrict__ Whi,
        const __bf16* __restrict__ Wlo, const float* __restrict__ bias,
        float* __restrict__ Out, int NC,
        const float* __restrict__ Xres, const float* __restrict__ skip) {
    const int rt   = blockIdx.x;
    const int lane = threadIdx.x & 31;
    const int wave = threadIdx.x >> 5;
    const int rowBase = rt * 16;

    __shared__ float xs[16 * DM_];
    for (int i = threadIdx.x; i < 16 * (DM_ / 4); i += 128) {
        int r = i >> 5, c4 = i & 31;
        const float4* src = reinterpret_cast<const float4*>(X + (size_t)(rowBase + r) * DM_);
        reinterpret_cast<float4*>(xs)[r * 32 + c4] = src[c4];
    }
    __syncthreads();

    // A 16x32 bf16 fragment: lane holds row (lane&15); kbase=(lane>>4)*8;
    // slots 0..7 -> K=kbase+j, slots 8..15 -> K=16+kbase+j (within each K32 chunk)
    const int ar  = lane & 15;
    const int akb = (lane >> 4) * 8;
    v16bf ahi[4], alo[4];
#pragma unroll
    for (int c = 0; c < 4; ++c) {
#pragma unroll
        for (int j = 0; j < 8; ++j) {
            float f0 = xs[ar * DM_ + c * 32 + akb + j];
            __bf16 h0 = (__bf16)f0;
            ahi[c][j] = h0;
            alo[c][j] = (__bf16)(f0 - (float)h0);
            float f1 = xs[ar * DM_ + c * 32 + akb + 16 + j];
            __bf16 h1 = (__bf16)f1;
            ahi[c][8 + j] = h1;
            alo[c][8 + j] = (__bf16)(f1 - (float)h1);
        }
    }

    const int NT = NC >> 4;
    float al = 0.f, om = 0.f;
    if (skip) { float s = skip[0]; al = 1.f / (1.f + __expf(-s)); om = 1.f - al; }

    const v16bf* bh = reinterpret_cast<const v16bf*>(Whi);
    const v16bf* bl = reinterpret_cast<const v16bf*>(Wlo);

    for (int nt = wave; nt < NT; nt += 4) {
        v8f acc = {};
#pragma unroll
        for (int c = 0; c < 4; ++c) {
            size_t bidx = (size_t)(c * NT + nt) * 32 + lane;
            v16bf bhi = bh[bidx];
            v16bf blo = bl[bidx];
            acc = __builtin_amdgcn_wmma_f32_16x16x32_bf16(false, ahi[c], false, bhi,
                                                          (short)0, acc, false, false);
            acc = __builtin_amdgcn_wmma_f32_16x16x32_bf16(false, ahi[c], false, blo,
                                                          (short)0, acc, false, false);
            acc = __builtin_amdgcn_wmma_f32_16x16x32_bf16(false, alo[c], false, bhi,
                                                          (short)0, acc, false, false);
        }
        // C/D layout: lane col = lane&15; VGPR r -> row r + 8*(lane>>4)
        const int col  = nt * 16 + (lane & 15);
        const int moff = (lane >> 4) * 8;
        const float bcol = bias[col];
#pragma unroll
        for (int r8 = 0; r8 < 8; ++r8) {
            size_t row = (size_t)(rowBase + moff + r8);
            float v = acc[r8] + bcol;
            if (skip) v = al * v + om * Xres[row * DM_ + col];
            Out[row * NC + col] = v;
        }
    }
}

// ----------------------------------------------------------- edge pass 1
__global__ void k_edge_logits(const float* __restrict__ qkv_src,
                              const float* __restrict__ qkv_dst,
                              const int* __restrict__ src, const int* __restrict__ dst,
                              const float* __restrict__ p_rel,
                              float* __restrict__ lg, float* __restrict__ mx, int E) {
    int t = blockIdx.x * blockDim.x + threadIdx.x;
    if (t >= E * HH_) return;
    int e = t >> 3, h = t & 7;
    int s = src[e], d = dst[e];
    const float4* kv = reinterpret_cast<const float4*>(qkv_src + (size_t)s * NQKV_ + DM_ + h * DD_);
    const float4* qv = reinterpret_cast<const float4*>(qkv_dst + (size_t)d * NQKV_ + h * DD_);
    float acc = 0.f;
#pragma unroll
    for (int i = 0; i < 4; ++i) {
        float4 kk = kv[i], qq = qv[i];
        acc += kk.x * qq.x + kk.y * qq.y + kk.z * qq.z + kk.w * qq.w;
    }
    acc *= p_rel[h] * 0.25f;   // 1/sqrt(D), D=16
    lg[t] = acc;
    atomicMaxF(&mx[(size_t)d * HH_ + h], acc);
}

// ----------------------------------------------------------- edge pass 2
__global__ void k_edge_exp(float* __restrict__ lg, const int* __restrict__ dst,
                           const float* __restrict__ mx, float* __restrict__ den, int E) {
    int t = blockIdx.x * blockDim.x + threadIdx.x;
    if (t >= E * HH_) return;
    int e = t >> 3, h = t & 7;
    int d = dst[e];
    float ex = __expf(lg[t] - mx[(size_t)d * HH_ + h]);
    lg[t] = ex;
    atomicAdd(&den[(size_t)d * HH_ + h], ex);
}

// ----------------------------------------------------------- edge pass 3
__global__ void k_edge_scatter(const float* __restrict__ lg, const int* __restrict__ src,
                               const int* __restrict__ dst, const float* __restrict__ qkv_src,
                               const float* __restrict__ den, float* __restrict__ agg, int E) {
    int t = blockIdx.x * blockDim.x + threadIdx.x;
    if (t >= E * HH_) return;
    int e = t >> 3, h = t & 7;
    int s = src[e], d = dst[e];
    float a = lg[t] / (den[(size_t)d * HH_ + h] + 1e-16f);
    const float* v = qkv_src + (size_t)s * NQKV_ + 2 * DM_ + h * DD_;
    float* o = agg + (size_t)d * DM_ + h * DD_;
#pragma unroll
    for (int i = 0; i < DD_; ++i) atomicAdd(&o[i], v[i] * a);
}

// ---------------------------------------------------------------- launcher
extern "C" void kernel_launch(void* const* d_in, const int* in_sizes, int n_in,
                              void* d_out, int out_size, void* d_ws, size_t ws_size,
                              hipStream_t stream) {
    (void)in_sizes; (void)n_in; (void)out_size; (void)ws_size;
    const float* x_p   = (const float*)d_in[0];
    const float* x_a   = (const float*)d_in[1];
    const int* src_ap  = (const int*)d_in[2];
    const int* dst_ap  = (const int*)d_in[3];
    const int* src_pa  = (const int*)d_in[4];
    const int* dst_pa  = (const int*)d_in[5];
    const float* Wq_p  = (const float*)d_in[6];  const float* bq_p = (const float*)d_in[7];
    const float* Wk_p  = (const float*)d_in[8];  const float* bk_p = (const float*)d_in[9];
    const float* Wv_p  = (const float*)d_in[10]; const float* bv_p = (const float*)d_in[11];
    const float* Wa_p  = (const float*)d_in[12]; const float* ba_p = (const float*)d_in[13];
    const float* sk_p  = (const float*)d_in[14];
    const float* Wq_a  = (const float*)d_in[15]; const float* bq_a = (const float*)d_in[16];
    const float* Wk_a  = (const float*)d_in[17]; const float* bk_a = (const float*)d_in[18];
    const float* Wv_a  = (const float*)d_in[19]; const float* bv_a = (const float*)d_in[20];
    const float* Wa_a  = (const float*)d_in[21]; const float* ba_a = (const float*)d_in[22];
    const float* sk_a  = (const float*)d_in[23];
    const float* a_ap  = (const float*)d_in[24]; const float* m_ap = (const float*)d_in[25];
    const float* p_ap  = (const float*)d_in[26];
    const float* a_pa  = (const float*)d_in[27]; const float* m_pa = (const float*)d_in[28];
    const float* p_pa  = (const float*)d_in[29];

    char* base = (char*)d_ws;
    size_t off = 0;
    auto take = [&](size_t bytes) -> char* {
        char* p = base + off;
        off = (off + bytes + 255) & ~(size_t)255;
        return p;
    };
    float*  qkv_p  = (float*)take((size_t)NP_ * NQKV_ * 4);
    float*  qkv_a  = (float*)take((size_t)NA_ * NQKV_ * 4);
    float*  agg_p  = (float*)take((size_t)NP_ * DM_ * 4);
    float*  agg_a  = (float*)take((size_t)NA_ * DM_ * 4);
    float*  lg_ap  = (float*)take((size_t)EE_ * HH_ * 4);
    float*  lg_pa  = (float*)take((size_t)EE_ * HH_ * 4);
    float*  mx_p   = (float*)take((size_t)NP_ * HH_ * 4);
    float*  dn_p   = (float*)take((size_t)NP_ * HH_ * 4);
    float*  mx_a   = (float*)take((size_t)NA_ * HH_ * 4);
    float*  dn_a   = (float*)take((size_t)NA_ * HH_ * 4);
    float*  w3_p   = (float*)take((size_t)DM_ * NQKV_ * 4);
    float*  b3_p   = (float*)take(NQKV_ * 4);
    float*  w3_a   = (float*)take((size_t)DM_ * NQKV_ * 4);
    float*  b3_a   = (float*)take(NQKV_ * 4);
    __bf16* w3hi_p = (__bf16*)take((size_t)DM_ * NQKV_ * 2);
    __bf16* w3lo_p = (__bf16*)take((size_t)DM_ * NQKV_ * 2);
    __bf16* w3hi_a = (__bf16*)take((size_t)DM_ * NQKV_ * 2);
    __bf16* w3lo_a = (__bf16*)take((size_t)DM_ * NQKV_ * 2);
    __bf16* wahi_p = (__bf16*)take((size_t)DM_ * DM_ * 2);
    __bf16* walo_p = (__bf16*)take((size_t)DM_ * DM_ * 2);
    __bf16* wahi_a = (__bf16*)take((size_t)DM_ * DM_ * 2);
    __bf16* walo_a = (__bf16*)take((size_t)DM_ * DM_ * 2);

    // 1) fused weights: papers are sources of "pa" edges, authors of "ap"
    k_build_w3<<<NQKV_, 128, 0, stream>>>(Wq_p, bq_p, Wk_p, bk_p, Wv_p, bv_p, a_pa, m_pa, w3_p, b3_p);
    k_build_w3<<<NQKV_, 128, 0, stream>>>(Wq_a, bq_a, Wk_a, bk_a, Wv_a, bv_a, a_ap, m_ap, w3_a, b3_a);

    // 2) pack to WMMA B-fragment bf16 hi/lo
    int pt3 = (DM_ * NQKV_ + 255) / 256;
    int pt1 = (DM_ * DM_ + 255) / 256;
    k_pack_w<<<pt3, 256, 0, stream>>>(w3_p, NQKV_, w3hi_p, w3lo_p);
    k_pack_w<<<pt3, 256, 0, stream>>>(w3_a, NQKV_, w3hi_a, w3lo_a);
    k_pack_w<<<pt1, 256, 0, stream>>>(Wa_p, DM_, wahi_p, walo_p);
    k_pack_w<<<pt1, 256, 0, stream>>>(Wa_a, DM_, wahi_a, walo_a);

    // 3) fused Q|K_rel|V_rel GEMMs (bf16x3 WMMA)
    k_gemm_bf16x3<<<NP_ / 16, 128, 0, stream>>>(x_p, w3hi_p, w3lo_p, b3_p, qkv_p, NQKV_, nullptr, nullptr);
    k_gemm_bf16x3<<<NA_ / 16, 128, 0, stream>>>(x_a, w3hi_a, w3lo_a, b3_a, qkv_a, NQKV_, nullptr, nullptr);

    // 4) init segment buffers
    k_fill<<<(NP_ * HH_ + 255) / 256, 256, 0, stream>>>(mx_p, -INFINITY, NP_ * HH_);
    k_fill<<<(NP_ * HH_ + 255) / 256, 256, 0, stream>>>(dn_p, 0.f, NP_ * HH_);
    k_fill<<<(NP_ * DM_ + 255) / 256, 256, 0, stream>>>(agg_p, 0.f, NP_ * DM_);
    k_fill<<<(NA_ * HH_ + 255) / 256, 256, 0, stream>>>(mx_a, -INFINITY, NA_ * HH_);
    k_fill<<<(NA_ * HH_ + 255) / 256, 256, 0, stream>>>(dn_a, 0.f, NA_ * HH_);
    k_fill<<<(NA_ * DM_ + 255) / 256, 256, 0, stream>>>(agg_a, 0.f, NA_ * DM_);

    // 5-7) edge passes (segment softmax + message scatter)
    int eg = (EE_ * HH_ + 255) / 256;
    k_edge_logits<<<eg, 256, 0, stream>>>(qkv_a, qkv_p, src_ap, dst_ap, p_ap, lg_ap, mx_p, EE_);
    k_edge_logits<<<eg, 256, 0, stream>>>(qkv_p, qkv_a, src_pa, dst_pa, p_pa, lg_pa, mx_a, EE_);
    k_edge_exp<<<eg, 256, 0, stream>>>(lg_ap, dst_ap, mx_p, dn_p, EE_);
    k_edge_exp<<<eg, 256, 0, stream>>>(lg_pa, dst_pa, mx_a, dn_a, EE_);
    k_edge_scatter<<<eg, 256, 0, stream>>>(lg_ap, src_ap, dst_ap, qkv_a, dn_p, agg_p, EE_);
    k_edge_scatter<<<eg, 256, 0, stream>>>(lg_pa, src_pa, dst_pa, qkv_p, dn_a, agg_a, EE_);

    // 8) output projection + sigmoid skip blend (fused epilogue)
    float* out = (float*)d_out;
    k_gemm_bf16x3<<<NP_ / 16, 128, 0, stream>>>(agg_p, wahi_p, walo_p, ba_p, out, DM_, x_p, sk_p);
    k_gemm_bf16x3<<<NA_ / 16, 128, 0, stream>>>(agg_a, wahi_a, walo_a, ba_a,
                                                out + (size_t)NP_ * DM_, DM_, x_a, sk_a);
}
